// LSTMNeuralNetwork_61572651155881
// MI455X (gfx1250) — matrix operand order repaired and checked by reading, not compiled, not run on hardware
//
#include <hip/hip_runtime.h>
#include <math.h>

// ---------------------------------------------------------------------------
// 2-layer LSTM (H=300) + MLP head for MI455X (gfx1250), wave32 + WMMA.
//
// Per-step fused kernel: gates = [h | x_t] @ [Whh^T ; Wih^T] via
// v_wmma_f32_16x16x32_f16, with the gate dimension PERMUTED as n' = 4*j+gate
// so each workgroup holds all four gates of its hidden units; the LSTM cell
// update happens in the same kernel via an LDS exchange (no gates buffer,
// one launch per timestep).
// ---------------------------------------------------------------------------

typedef __attribute__((ext_vector_type(16))) _Float16 v16h;
typedef __attribute__((ext_vector_type(8)))  float    v8f;

#define HS   300          // hidden size
#define HP   320          // hidden padded to k-tile multiple
#define NGP  1280         // permuted/padded gate width: 4*HP
#define NTP  80           // n'-tiles (1280/16)
#define BB   256          // batch
#define TT   256          // time steps
#define KT_H 10           // k-tiles for hidden (padded 300 -> 320)
#define PACK_T_STRIDE (16*KT_H*32*16)   // halves per timestep of packed h (81920)
#define PACK_X_STRIDE (16*1*32*16)      // halves per timestep of packed x (8192)

// ---- A-layout mapping: element (M in 0..15, K in 0..31) of a 16x32 f16 tile
__device__ __forceinline__ void a_layout(int M, int K, int& lane, int& half) {
    int group = K >> 3;                 // 0..3
    lane = M + 16 * (group & 1);
    half = (K & 7) + 8 * (group >> 1);
}

__device__ __forceinline__ float sigmoidf_(float x) { return 1.0f / (1.0f + expf(-x)); }

// ---------------------------------------------------------------------------
// Pack W (row-major [1200, Kdim], rows ordered gate*300+j) into B-matrix lane
// layout with PERMUTED columns n' = 4*jp + gate (jp in [0,320), zero-padded).
// Element order matches GEMM load: (((kt*NTP + ntile)*32 + lane)*16 + half).
// B lane layout: lanes 0-15 hold K=0..7,16..23; lanes 16-31 hold K=8..15,24..31.
// ---------------------------------------------------------------------------
__global__ void pack_B_kernel(const float* __restrict__ W, int Kdim,
                              _Float16* __restrict__ Bp, int ktSec) {
    int idx = blockIdx.x * blockDim.x + threadIdx.x;
    int total = ktSec * NTP * 512;
    if (idx >= total) return;
    int kt    = idx / (NTP * 512);
    int rem   = idx % (NTP * 512);
    int ntile = rem / 512;
    int r2    = rem % 512;
    int lane  = r2 / 16;
    int half  = r2 % 16;
    int K  = kt * 32 + (half & 7) + 8 * (lane >= 16) + 16 * (half >> 3);
    int Np = ntile * 16 + (lane & 15);   // permuted column
    int jp = Np >> 2;
    int gate = Np & 3;
    float v = (K < Kdim && jp < HS) ? W[((size_t)gate * HS + jp) * Kdim + K] : 0.0f;
    Bp[idx] = (_Float16)v;
}

// Permuted combined bias: biasP[4*jp+gate] = bih[gate*300+jp]+bhh[...] (pad 0)
__global__ void bias_perm_kernel(const float* __restrict__ bih,
                                 const float* __restrict__ bhh,
                                 float* __restrict__ biasP) {
    int n = blockIdx.x * blockDim.x + threadIdx.x;
    if (n >= NGP) return;
    int jp = n >> 2, gate = n & 3;
    biasP[n] = (jp < HS) ? (bih[gate * HS + jp] + bhh[gate * HS + jp]) : 0.0f;
}

// Pack x [B,T,16] fp32 into per-t A-layout tiles (K padded 16 -> 32).
__global__ void pack_X0_kernel(const float* __restrict__ x, _Float16* __restrict__ Xp) {
    int idx = blockIdx.x * blockDim.x + threadIdx.x;
    if (idx >= TT * BB * 32) return;
    int t  = idx / (BB * 32);
    int r  = idx % (BB * 32);
    int b  = r / 32;
    int kp = r % 32;
    float v = (kp < 16) ? x[((size_t)b * TT + t) * 16 + kp] : 0.0f;
    int mtile = b >> 4, M = b & 15, lane, half;
    a_layout(M, kp, lane, half);
    Xp[(size_t)t * PACK_X_STRIDE + ((size_t)mtile * 32 + lane) * 16 + half] = (_Float16)v;
}

__global__ void zero_f_kernel(float* p, int n) {
    int i = blockIdx.x * blockDim.x + threadIdx.x;
    if (i < n) p[i] = 0.0f;
}
__global__ void zero_h_kernel(_Float16* p, int n) {
    int i = blockIdx.x * blockDim.x + threadIdx.x;
    if (i < n) p[i] = (_Float16)0.0f;
}

// ---------------------------------------------------------------------------
// Fused LSTM step: per-timestep GEMM (WMMA) + cell update via LDS.
//
// Ah: packed A tiles [16 mtiles][ktA][32 lanes] (v16h each) -- previous h
// Ax: packed A tiles [16 mtiles][ktX][32 lanes]            -- step input
// Bp: packed B tiles [(ktA+ktX)][NTP][32 lanes], permuted columns n'=4*j+gate
//
// Block = 256 threads (8 waves); grid = (10, 16): blockIdx.x covers 8 n'-tiles
// = 128 n' = 32 hidden units; blockIdx.y = m-tile (16 batches). Grid covers
// tiles exactly, so EXEC is all-ones through every WMMA.
// ---------------------------------------------------------------------------
__global__ __launch_bounds__(256)
void lstm_step_fused(const v16h* __restrict__ Ah, int ktA,
                     const v16h* __restrict__ Ax, int ktX,
                     const v16h* __restrict__ Bp,
                     const float* __restrict__ biasP,   // [1280] permuted
                     float* __restrict__ c,             // [256][300]
                     _Float16* __restrict__ hpacked_out,// per-t A layout
                     float* __restrict__ hrow_seq,      // nullable [B*T,300]
                     int t) {
    __shared__ float Cs[8][16][16];    // 8 waves x 16x16 f32 tiles (8 KB)

    const int lane = threadIdx.x & 31;
    const int wave = threadIdx.x >> 5;
    const int ntp  = blockIdx.x * 8 + wave;   // 0..79 (n'-tile)
    const int mtile = blockIdx.y;             // 0..15

    v8f acc = {};
    for (int kt = 0; kt < ktA; ++kt) {
        v16h a = Ah[((size_t)mtile * ktA + kt) * 32 + lane];
        v16h b = Bp[((size_t)kt * NTP + ntp) * 32 + lane];
        acc = __builtin_amdgcn_wmma_f32_16x16x32_f16(false, a, false, b,
                                                     (short)0, acc, false, false);
    }
    for (int kx = 0; kx < ktX; ++kx) {
        v16h a = Ax[((size_t)mtile * ktX + kx) * 32 + lane];
        v16h b = Bp[((size_t)(ktA + kx) * NTP + ntp) * 32 + lane];
        acc = __builtin_amdgcn_wmma_f32_16x16x32_f16(false, a, false, b,
                                                     (short)0, acc, false, false);
    }

    // C/D layout: VGPR r, lane l -> M = r + 8*(l>=16), N = l&15
    const int mhi = (lane >= 16) ? 8 : 0;
    const int col = lane & 15;
#pragma unroll
    for (int r = 0; r < 8; ++r)
        Cs[wave][r + mhi][col] = acc[r];
    __syncthreads();

    // Cell update: this block owns batches [mtile*16, +16) x hidden
    // [blockIdx.x*32, +32). 512 elements, 256 threads -> 2 each.
    for (int e = threadIdx.x; e < 512; e += 256) {
        int M  = e >> 5;                    // batch within tile
        int jl = e & 31;                    // hidden unit within block
        int b  = mtile * 16 + M;
        int jp = blockIdx.x * 32 + jl;      // 0..319 (padded hidden index)
        float h = 0.0f;
        if (jp < HS) {
            int n0 = 4 * jl;                // local permuted column of gate i
            float gi = Cs[(n0 + 0) >> 4][M][(n0 + 0) & 15] + biasP[4 * jp + 0];
            float gf = Cs[(n0 + 1) >> 4][M][(n0 + 1) & 15] + biasP[4 * jp + 1];
            float gg = Cs[(n0 + 2) >> 4][M][(n0 + 2) & 15] + biasP[4 * jp + 2];
            float go = Cs[(n0 + 3) >> 4][M][(n0 + 3) & 15] + biasP[4 * jp + 3];
            float iv = sigmoidf_(gi);
            float fv = sigmoidf_(gf);
            float gv = tanhf(gg);
            float ov = sigmoidf_(go);
            float cc = fv * c[(size_t)b * HS + jp] + iv * gv;
            c[(size_t)b * HS + jp] = cc;
            h = ov * tanhf(cc);
            if (hrow_seq) hrow_seq[((size_t)b * TT + t) * HS + jp] = h;
        }
        int kt = jp >> 5, K = jp & 31, lane2, half;
        a_layout(M, K, lane2, half);
        hpacked_out[(((size_t)mtile * KT_H + kt) * 32 + lane2) * 16 + half] = (_Float16)h;
    }
}

// ---------------------------------------------------------------------------
// MLP head: tanh(h@W1^T+b1) -> tanh(@W2^T+b2) -> sigmoid(@W3^T+b3)
// One thread per (b,t) row; weights are tiny and L2-resident.
// ---------------------------------------------------------------------------
__global__ void head_kernel(const float* __restrict__ Hseq,
                            const float* __restrict__ W1, const float* __restrict__ b1,
                            const float* __restrict__ W2, const float* __restrict__ b2,
                            const float* __restrict__ W3, const float* __restrict__ b3,
                            float* __restrict__ out) {
    int row = blockIdx.x * blockDim.x + threadIdx.x;
    if (row >= BB * TT) return;
    const float* h = Hseq + (size_t)row * HS;
    float t1[30];
    for (int j = 0; j < 30; ++j) {
        float s = b1[j];
        const float* w = W1 + (size_t)j * HS;
        for (int k = 0; k < HS; ++k) s += h[k] * w[k];
        t1[j] = tanhf(s);
    }
    float o = b3[0];
    for (int j2 = 0; j2 < 100; ++j2) {
        float s = b2[j2];
        const float* w = W2 + (size_t)j2 * 30;
        for (int j = 0; j < 30; ++j) s += t1[j] * w[j];
        o += tanhf(s) * W3[j2];
    }
    out[row] = sigmoidf_(o);
}

// ---------------------------------------------------------------------------
static inline size_t align256(size_t x) { return (x + 255) & ~(size_t)255; }

extern "C" void kernel_launch(void* const* d_in, const int* in_sizes, int n_in,
                              void* d_out, int out_size, void* d_ws, size_t ws_size,
                              hipStream_t stream) {
    const float* x    = (const float*)d_in[0];
    const float* Wih0 = (const float*)d_in[1];
    const float* Whh0 = (const float*)d_in[2];
    const float* bih0 = (const float*)d_in[3];
    const float* bhh0 = (const float*)d_in[4];
    const float* Wih1 = (const float*)d_in[5];
    const float* Whh1 = (const float*)d_in[6];
    const float* bih1 = (const float*)d_in[7];
    const float* bhh1 = (const float*)d_in[8];
    const float* W1   = (const float*)d_in[9];
    const float* b1   = (const float*)d_in[10];
    const float* W2   = (const float*)d_in[11];
    const float* b2   = (const float*)d_in[12];
    const float* W3   = (const float*)d_in[13];
    const float* b3   = (const float*)d_in[14];
    float* out = (float*)d_out;

    // ---- carve workspace (~128 MB total) ----
    char* ws = (char*)d_ws;
    size_t off = 0;
    auto alloc = [&](size_t bytes) -> char* {
        char* p = ws + off;
        off = align256(off + bytes);
        return p;
    };
    _Float16* B0p   = (_Float16*)alloc((size_t)11 * NTP * 512 * 2);         // layer0 B (h + x sections)
    _Float16* B1p   = (_Float16*)alloc((size_t)20 * NTP * 512 * 2);         // layer1 B (h + h0 sections)
    _Float16* X0p   = (_Float16*)alloc((size_t)TT * PACK_X_STRIDE * 2);     // packed x per t
    _Float16* H0p   = (_Float16*)alloc((size_t)TT * PACK_T_STRIDE * 2);     // packed layer0 h per t
    _Float16* h1buf = (_Float16*)alloc((size_t)2 * PACK_T_STRIDE * 2);      // layer1 h double buffer
    _Float16* zeroP = (_Float16*)alloc((size_t)PACK_T_STRIDE * 2);          // packed zero state
    float*    cbuf  = (float*)alloc((size_t)BB * HS * 4);
    float*    bias0 = (float*)alloc((size_t)NGP * 4);
    float*    bias1 = (float*)alloc((size_t)NGP * 4);
    float*    H1seq = (float*)alloc((size_t)BB * TT * HS * 4);
    (void)ws_size;

    const int TPB = 256;

    // ---- one-time (per call) packing ----
    pack_B_kernel<<<(10 * NTP * 512 + TPB - 1) / TPB, TPB, 0, stream>>>(Whh0, HS, B0p, 10);
    pack_B_kernel<<<( 1 * NTP * 512 + TPB - 1) / TPB, TPB, 0, stream>>>(Wih0, 16, B0p + (size_t)10 * NTP * 512, 1);
    pack_B_kernel<<<(10 * NTP * 512 + TPB - 1) / TPB, TPB, 0, stream>>>(Whh1, HS, B1p, 10);
    pack_B_kernel<<<(10 * NTP * 512 + TPB - 1) / TPB, TPB, 0, stream>>>(Wih1, HS, B1p + (size_t)10 * NTP * 512, 10);
    bias_perm_kernel<<<(NGP + TPB - 1) / TPB, TPB, 0, stream>>>(bih0, bhh0, bias0);
    bias_perm_kernel<<<(NGP + TPB - 1) / TPB, TPB, 0, stream>>>(bih1, bhh1, bias1);
    pack_X0_kernel<<<(TT * BB * 32 + TPB - 1) / TPB, TPB, 0, stream>>>(x, X0p);
    zero_h_kernel<<<(PACK_T_STRIDE + TPB - 1) / TPB, TPB, 0, stream>>>(zeroP, PACK_T_STRIDE);

    dim3 ggrid(10, 16);        // 80 n'-tiles x 16 m-tiles, 8 waves/block

    // ---- layer 0 recurrence ----
    zero_f_kernel<<<(BB * HS + TPB - 1) / TPB, TPB, 0, stream>>>(cbuf, BB * HS);
    for (int t = 0; t < TT; ++t) {
        const v16h* Ah = (const v16h*)((t == 0) ? zeroP : H0p + (size_t)(t - 1) * PACK_T_STRIDE);
        const v16h* Ax = (const v16h*)(X0p + (size_t)t * PACK_X_STRIDE);
        lstm_step_fused<<<ggrid, TPB, 0, stream>>>(Ah, KT_H, Ax, 1, (const v16h*)B0p,
                                                   bias0, cbuf,
                                                   H0p + (size_t)t * PACK_T_STRIDE,
                                                   nullptr, t);
    }

    // ---- layer 1 recurrence ----
    zero_f_kernel<<<(BB * HS + TPB - 1) / TPB, TPB, 0, stream>>>(cbuf, BB * HS);
    for (int t = 0; t < TT; ++t) {
        const v16h* Ah = (const v16h*)((t == 0) ? zeroP : h1buf + (size_t)((t - 1) & 1) * PACK_T_STRIDE);
        const v16h* Ax = (const v16h*)(H0p + (size_t)t * PACK_T_STRIDE);
        lstm_step_fused<<<ggrid, TPB, 0, stream>>>(Ah, KT_H, Ax, KT_H, (const v16h*)B1p,
                                                   bias1, cbuf,
                                                   h1buf + (size_t)(t & 1) * PACK_T_STRIDE,
                                                   H1seq, t);
    }

    // ---- MLP head ----
    head_kernel<<<(BB * TT + TPB - 1) / TPB, TPB, 0, stream>>>(H1seq, W1, b1, W2, b2, W3, b3, out);
}
